// EnergySharingPredictor_77592879169751
// MI455X (gfx1250) — matrix operand order repaired and checked by reading, not compiled, not run on hardware
//
#include <hip/hip_runtime.h>
#include <hip/hip_bf16.h>

typedef __attribute__((ext_vector_type(16))) _Float16 v16h;
typedef __attribute__((ext_vector_type(8)))  float    v8f;
typedef __attribute__((ext_vector_type(4)))  unsigned u32x4;
typedef __attribute__((ext_vector_type(8)))  int      i32x8;
typedef __attribute__((ext_vector_type(4)))  int      i32x4;

#define NB   4096
#define NCL  128
#define KPC  32
#define EDIM 128
#define TDIM 24
constexpr size_t NNsz = (size_t)NB * (size_t)NB;

union VF8  { v8f  v; float    f[8];  };
union VH16 { v16h v; _Float16 h[16]; };

// ---------------------------------------------------------------------------
// Kernel 1: stable rank within cluster + net = gen - cons at current hour
// ---------------------------------------------------------------------------
__global__ void prep_kernel(const int* __restrict__ assign,
                            const float* __restrict__ gen,
                            const float* __restrict__ cons,
                            const int* __restrict__ hourp,
                            int* __restrict__ cidx, int* __restrict__ krank,
                            int* __restrict__ gmap, float* __restrict__ netArr)
{
  int i = blockIdx.x * blockDim.x + threadIdx.x;
  if (i >= NB) return;
  int c = assign[i];
  int r = 0;
  for (int j = 0; j < i; ++j) r += (assign[j] == c) ? 1 : 0;
  cidx[i] = c; krank[i] = r;
  gmap[c * KPC + r] = i;
  int h = hourp[0];
  netArr[i] = gen[i * TDIM + h] - cons[i * TDIM + h];
}

// ---------------------------------------------------------------------------
// Kernel 2: f32 -> f16 conversions / weight packing into WMMA B layouts
//   bigB[e][f], f in [0,128)=W1a, [128,256)=W1b, [256,288)=ps_w1
//   B2h [e][f] = fp_w2[f][e]  (128 x 64)
// ---------------------------------------------------------------------------
__global__ void cvt_kernel(const float* __restrict__ emb, const float* __restrict__ fp_w1,
                           const float* __restrict__ ps_w1, const float* __restrict__ fp_w2,
                           _Float16* __restrict__ emb_h, _Float16* __restrict__ bigB,
                           _Float16* __restrict__ B2h)
{
  int idx = blockIdx.x * blockDim.x + threadIdx.x;
  if (idx < NB * EDIM) { emb_h[idx] = (_Float16)emb[idx]; return; }
  idx -= NB * EDIM;
  if (idx < 128 * 288) {
    int e = idx / 288, f = idx % 288;
    float v;
    if (f < 128)       v = fp_w1[f * 258 + e];
    else if (f < 256)  v = fp_w1[(f - 128) * 258 + 128 + e];
    else               v = ps_w1[(f - 256) * 128 + e];
    bigB[idx] = (_Float16)v;
    return;
  }
  idx -= 128 * 288;
  if (idx < 128 * 64) {
    int e = idx / 64, f = idx % 64;
    B2h[idx] = (_Float16)fp_w2[f * 128 + e];
  }
}

// ---------------------------------------------------------------------------
// WMMA fragment loaders (CDNA5 16x16x32 f16 layouts, wave32)
// ---------------------------------------------------------------------------
__device__ __forceinline__ v16h load_a_rowmajor(const _Float16* A, int lda, int m0, int k0)
{
  int lane = threadIdx.x & 31;
  int row  = m0 + (lane & 15);
  int kb   = k0 + ((lane & 16) ? 8 : 0);
  const _Float16* p = A + (size_t)row * lda + kb;
  VH16 a;
#pragma unroll
  for (int h = 0; h < 8; ++h) { a.h[h] = p[h]; a.h[h + 8] = p[h + 16]; }
  return a.v;
}
__device__ __forceinline__ v16h load_b_rowmajor(const _Float16* B, int ldb, int k0, int n0)
{
  int lane = threadIdx.x & 31;
  const _Float16* p = B + (size_t)(k0 + ((lane & 16) ? 8 : 0)) * ldb + n0 + (lane & 15);
  VH16 b;
#pragma unroll
  for (int h = 0; h < 8; ++h) { b.h[h] = p[(size_t)h * ldb]; b.h[h + 8] = p[(size_t)(h + 16) * ldb]; }
  return b.v;
}

// ---------------------------------------------------------------------------
// TDM: DMA a contiguous 16KB block (128x64 f16) from global into LDS.
// D# per CDNA5 ISA ch.8: group0 = {count/type/addr}, group1 = dims/strides.
// 1-row tile: tile_dim0 = 8192 elements, data_size = 2 bytes.
// LDS offset = low 32 bits of the flat shared-aperture pointer (ISA 10.2).
// ---------------------------------------------------------------------------
#if __has_builtin(__builtin_amdgcn_tensor_load_to_lds)
#define HAVE_TDM 1
__device__ __forceinline__ void tdm_load_16k(_Float16* ldsDst, const _Float16* gsrc)
{
  unsigned lds = (unsigned)(unsigned long long)(size_t)ldsDst;
  unsigned long long ga = (unsigned long long)(size_t)gsrc;
  u32x4 g0 = { 1u,                                   // count=1, user descriptor
               lds,                                  // lds_addr
               (unsigned)(ga & 0xffffffffull),       // global_addr[31:0]
               (unsigned)((ga >> 32) & 0x1ffffffull) | (2u << 30) }; // addr[56:32] | type=2
  i32x8 g1 = { (int)(1u << 16),        // data_size=1 (2B), wg_mask=0
               (int)(0x2000u << 16),   // tensor_dim0 = 8192 (bits[63:48])
               (int)(1u << 16),        // tensor_dim0 hi=0, tensor_dim1=1
               (int)(0x2000u << 16),   // tensor_dim1 hi=0, tile_dim0=8192
               1,                      // tile_dim1=1, tile_dim2=0
               0x2000,                 // tensor_dim0_stride = 8192
               (int)(0x2000u << 16),   // stride0 hi=0, tensor_dim1_stride=8192
               0 };
  i32x4 gz = { 0, 0, 0, 0 };
#if defined(__clang_major__) && (__clang_major__ >= 23)
  i32x8 gz8 = { 0, 0, 0, 0, 0, 0, 0, 0 };
  __builtin_amdgcn_tensor_load_to_lds(g0, g1, gz, gz, gz8, 0);
#else
  __builtin_amdgcn_tensor_load_to_lds(g0, g1, gz, gz, 0);
#endif
}
#endif

// ---------------------------------------------------------------------------
// Kernel 3: hcat = emb(4096x128) @ bigB(128x288) -> f32, via v_wmma f16
// ---------------------------------------------------------------------------
__global__ void __launch_bounds__(256) gemm_hcat_kernel(const _Float16* __restrict__ emb_h,
                                                        const _Float16* __restrict__ bigB,
                                                        float* __restrict__ hcat)
{
  int wave = threadIdx.x >> 5;
  int tile = blockIdx.x * 8 + wave;
  int mt = tile / 18, nt = tile % 18;
  VF8 acc;
#pragma unroll
  for (int e = 0; e < 8; ++e) acc.f[e] = 0.f;
#pragma unroll
  for (int kt = 0; kt < 4; ++kt) {
    v16h a = load_a_rowmajor(emb_h, EDIM, mt * 16, kt * 32);
    v16h b = load_b_rowmajor(bigB, 288, kt * 32, nt * 16);
    acc.v = __builtin_amdgcn_wmma_f32_16x16x32_f16(false, a, false, b, (short)0, acc.v, false, false);
  }
  int lane  = threadIdx.x & 31;
  int rbase = mt * 16 + ((lane & 16) ? 8 : 0);
  int col   = nt * 16 + (lane & 15);
#pragma unroll
  for (int r = 0; r < 8; ++r) hcat[(size_t)(rbase + r) * 288 + col] = acc.f[r];
}

// ---------------------------------------------------------------------------
// Kernel 4: per-cluster work (1 block = 1 cluster, 256 threads = 8 waves)
// ---------------------------------------------------------------------------
__global__ void __launch_bounds__(256) cluster_kernel(
    const int* __restrict__ gmap, const float* __restrict__ netArr,
    const float* __restrict__ hcat, const float* __restrict__ pos,
    const int* __restrict__ hourp,
    const float* __restrict__ ps_b1, const float* __restrict__ ps_w2, const float* __restrict__ ps_b2,
    const float* __restrict__ en_w1, const float* __restrict__ en_b1,
    const float* __restrict__ en_w2, const float* __restrict__ en_b2,
    const float* __restrict__ fp_w1, const float* __restrict__ fp_b1,
    const float* __restrict__ fp_b2, const float* __restrict__ fp_w3, const float* __restrict__ fp_b3,
    const _Float16* __restrict__ B2h,
    float* __restrict__ ws_flows, float* __restrict__ ws_eff, float* __restrict__ ws_totals,
    float* __restrict__ outSent, float* __restrict__ outRecv, float* __restrict__ outNet)
{
  __shared__ float    hiC[KPC * EDIM];
  __shared__ float    hjC[KPC * EDIM];
  __shared__ _Float16 B2s[EDIM * 64];
  __shared__ float    distF[KPC * KPC];   // reused as flows_o after GEMM stage
  __shared__ float    effS[KPC * KPC];
  __shared__ float    pfS[KPC * KPC];
  __shared__ float    cvec[EDIM], dvec[EDIM];
  __shared__ float    w3s[64], b2sh[64];
  __shared__ float    prioS[KPC], netS[KPC], posx[KPC], posy[KPC], dnetS[KPC];
  __shared__ int      gS[KPC], orderS[KPC];
  __shared__ unsigned actS[KPC];

  const int c   = blockIdx.x;
  const int tid = threadIdx.x;
  const float hourf = (float)hourp[0] / 24.0f;

#if defined(HAVE_TDM)
  if (tid < 32) tdm_load_16k(B2s, B2h);   // one TDM issue from wave 0 (EXEC-independent)
#endif

  if (tid < KPC) {
    int b = gmap[c * KPC + tid];
    gS[tid] = b;
    posx[tid] = pos[2 * b]; posy[tid] = pos[2 * b + 1];
    netS[tid] = netArr[b];
    float s = ps_b2[0];
    for (int f = 0; f < 32; ++f) {
      float h = hcat[(size_t)b * 288 + 256 + f] + ps_b1[f];
      s += fmaxf(h, 0.f) * ps_w2[f];
    }
    prioS[tid] = 1.f / (1.f + expf(-s));
  }
  if (tid < 128) {
    dvec[tid] = fp_w1[tid * 258 + 256];
    cvec[tid] = hourf * fp_w1[tid * 258 + 257] + fp_b1[tid];
  } else if (tid < 192) {
    int f = tid - 128;
    w3s[f]  = fp_w3[f];
    b2sh[f] = fp_b2[f];
  }
#if !defined(HAVE_TDM)
  for (int t = tid; t < EDIM * 64; t += 256) B2s[t] = B2h[t];
#endif

#if defined(HAVE_TDM)
  if (tid < 32) __builtin_amdgcn_s_wait_tensorcnt(0);
#endif
  __syncthreads();

  for (int t = tid; t < KPC * EDIM; t += 256) {
    int i = t >> 7, k = t & 127;
    size_t b = (size_t)gS[i] * 288;
    hiC[t] = hcat[b + k];
    hjC[t] = hcat[b + 128 + k];
  }
  for (int t = tid; t < KPC * KPC; t += 256) {
    int i = t >> 5, j = t & 31;
    float dx = posx[i] - posx[j], dy = posy[i] - posy[j];
    float d = sqrtf(dx * dx + dy * dy);
    distF[t] = d;
    float x = d * 0.001f;
    float s = en_b2[0];
#pragma unroll
    for (int k = 0; k < 16; ++k) {
      float eh = fmaxf(x * en_w1[k] + en_b1[k], 0.f);
      s += eh * en_w2[k];
    }
    effS[t] = 0.85f + 0.13f / (1.f + expf(-s));
  }
  __syncthreads();

  // --- h1 built on the fly in A-layout; h2 = h1 @ fp_w2^T via WMMA; pflow ---
  const int wave = tid >> 5, lane = tid & 31;
  const int li = lane & 15, hb = (lane >> 4) & 1;
  const float b3v = fp_b3[0];
  for (int mt = wave; mt < 64; mt += 8) {
    const int m0  = mt * 16;
    const int row = m0 + li;                 // pair index (i*32 + j)
    const int ai = row >> 5, aj = row & 31;
    const float dval = distF[ai * KPC + aj];
    VF8 acc[4];
#pragma unroll
    for (int nt = 0; nt < 4; ++nt)
#pragma unroll
      for (int e = 0; e < 8; ++e) acc[nt].f[e] = 0.f;
#pragma unroll
    for (int kt = 0; kt < 4; ++kt) {
      const int kb = kt * 32 + (hb ? 8 : 0);
      VH16 a;
#pragma unroll
      for (int h = 0; h < 8; ++h) {
        int K = kb + h;
        a.h[h]     = (_Float16)fmaxf(hiC[ai * EDIM + K] + hjC[aj * EDIM + K] + dval * dvec[K] + cvec[K], 0.f);
        K = kb + 16 + h;
        a.h[h + 8] = (_Float16)fmaxf(hiC[ai * EDIM + K] + hjC[aj * EDIM + K] + dval * dvec[K] + cvec[K], 0.f);
      }
#pragma unroll
      for (int nt = 0; nt < 4; ++nt) {
        VH16 bfr;
        const _Float16* bp = &B2s[(kt * 32 + (hb ? 8 : 0)) * 64 + nt * 16 + li];
#pragma unroll
        for (int h = 0; h < 8; ++h) { bfr.h[h] = bp[h * 64]; bfr.h[h + 8] = bp[(h + 16) * 64]; }
        acc[nt].v = __builtin_amdgcn_wmma_f32_16x16x32_f16(false, a.v, false, bfr.v, (short)0, acc[nt].v, false, false);
      }
    }
    // epilogue: relu(h2 + b2) . w3, reduce across 16-lane half, softplus
    float partial[8];
#pragma unroll
    for (int r = 0; r < 8; ++r) partial[r] = 0.f;
#pragma unroll
    for (int nt = 0; nt < 4; ++nt) {
      const int f = nt * 16 + li;
      const float wv = w3s[f], bv = b2sh[f];
#pragma unroll
      for (int r = 0; r < 8; ++r) partial[r] += fmaxf(acc[nt].f[r] + bv, 0.f) * wv;
    }
#pragma unroll
    for (int m = 1; m < 16; m <<= 1)
#pragma unroll
      for (int r = 0; r < 8; ++r) partial[r] += __shfl_xor(partial[r], m, 32);
    if (li == 0) {
      const int rb = m0 + (hb ? 8 : 0);
#pragma unroll
      for (int r = 0; r < 8; ++r) {
        float x = partial[r] + b3v;
        pfS[rb + r] = (x > 20.f) ? x : log1pf(expf(x));
      }
    }
  }
  __syncthreads();

  // --- greedy allocation (inherently sequential): lane 0 of the block ---
  if (tid == 0) {
    unsigned usedMask = 0;
    for (int oi = 0; oi < KPC; ++oi) {          // stable descending argsort
      int best = -1; float bp = -1e30f;
      for (int t = 0; t < KPC; ++t)
        if (!((usedMask >> t) & 1) && prioS[t] > bp) { bp = prioS[t]; best = t; }
      usedMask |= (1u << best);
      orderS[oi] = best;
    }
    float dn[KPC];
    for (int t = 0; t < KPC; ++t) dn[t] = netS[orderS[t]];
    for (int oi = 0; oi < KPC; ++oi) {
      const int i = orderS[oi];
      float avail = fmaxf(netS[i], 0.f);
      unsigned abits = 0;
      for (int oj = 0; oj < KPC; ++oj) {
        const int j = orderS[oj];
        const float needed = -dn[oj];
        float f = 0.f;
        if (avail > 0.f && needed > 0.f) {
          f = fminf(fminf(avail, needed), pfS[i * KPC + j]);
          abits |= (1u << oj);
        }
        dn[oj] += f * effS[i * KPC + j];
        avail  -= f;
        distF[oi * KPC + oj] = f;               // flows in order-space (dist dead now)
      }
      actS[oi] = abits;
    }
    for (int t = 0; t < KPC; ++t) dnetS[orderS[t]] = dn[t];
  }
  __syncthreads();

  // --- scatter block results + per-building sums ---
  for (int t = tid; t < KPC * KPC; t += 256) {
    int oi = t >> 5, oj = t & 31;
    int i = orderS[oi], j = orderS[oj];
    float f = distF[t];
    int a = (actS[oi] >> oj) & 1;
    size_t base = (size_t)c * (KPC * KPC) + (size_t)i * KPC + j;
    ws_flows[base] = f;
    ws_eff[base]   = a ? effS[i * KPC + j] : 1.0f;
  }
  if (tid < KPC) {
    float sent = 0.f;
    for (int oj = 0; oj < KPC; ++oj) sent += distF[tid * KPC + oj];
    outSent[gS[orderS[tid]]] = sent;
    float recv = 0.f;
    int j = orderS[tid];
    for (int oi = 0; oi < KPC; ++oi)
      recv += distF[oi * KPC + tid] * effS[orderS[oi] * KPC + j];
    outRecv[gS[j]] = recv;
    outNet[gS[tid]] = dnetS[tid];
  }
  if (tid == 0) {
    float tot = 0.f;
    for (int t = 0; t < KPC * KPC; ++t) tot += distF[t];
    ws_totals[c] = tot;
  }
}

// ---------------------------------------------------------------------------
// Kernel 5: stream the full 4096x4096 sharing + effmat. This is the roofline
// limiter (134 MB of writes @ 23.3 TB/s) -> use b128 stores, 4 elems/thread.
// ---------------------------------------------------------------------------
__global__ void __launch_bounds__(256) fill_kernel(const int* __restrict__ cidx,
                                                   const int* __restrict__ krank,
                                                   const float* __restrict__ ws_flows,
                                                   const float* __restrict__ ws_eff,
                                                   float* __restrict__ out)
{
  size_t idx4 = ((size_t)blockIdx.x * blockDim.x + threadIdx.x) * 4;
  int r  = (int)(idx4 >> 12);
  int q0 = (int)(idx4 & 4095);
  int cr = cidx[r];
  int kr = krank[r];
  float4 s = make_float4(0.f, 0.f, 0.f, 0.f);
  float4 e = make_float4(1.f, 1.f, 1.f, 1.f);
  float* sp = &s.x;
  float* ep = &e.x;
#pragma unroll
  for (int u = 0; u < 4; ++u) {
    int q = q0 + u;
    if (cidx[q] == cr) {
      size_t b = (size_t)cr * (KPC * KPC) + (size_t)kr * KPC + krank[q];
      sp[u] = ws_flows[b];
      ep[u] = ws_eff[b];
    }
  }
  *(float4*)(out + idx4)        = s;
  *(float4*)(out + NNsz + idx4) = e;
}

// ---------------------------------------------------------------------------
// Kernel 6: total_shared (fixed order -> deterministic)
// ---------------------------------------------------------------------------
__global__ void total_kernel(const float* __restrict__ ws_totals, float* __restrict__ out)
{
  if (threadIdx.x == 0 && blockIdx.x == 0) {
    float t = 0.f;
    for (int c = 0; c < NCL; ++c) t += ws_totals[c];
    out[2 * NNsz] = t;
  }
}

// ---------------------------------------------------------------------------
extern "C" void kernel_launch(void* const* d_in, const int* in_sizes, int n_in,
                              void* d_out, int out_size, void* d_ws, size_t ws_size,
                              hipStream_t stream)
{
  (void)in_sizes; (void)n_in; (void)out_size; (void)ws_size;
  const float* emb   = (const float*)d_in[0];
  const float* gen   = (const float*)d_in[1];
  const float* cons  = (const float*)d_in[2];
  const float* pos   = (const float*)d_in[3];
  const float* fp_w1 = (const float*)d_in[4];
  const float* fp_b1 = (const float*)d_in[5];
  const float* fp_w2 = (const float*)d_in[6];
  const float* fp_b2 = (const float*)d_in[7];
  const float* fp_w3 = (const float*)d_in[8];
  const float* fp_b3 = (const float*)d_in[9];
  const float* en_w1 = (const float*)d_in[10];
  const float* en_b1 = (const float*)d_in[11];
  const float* en_w2 = (const float*)d_in[12];
  const float* en_b2 = (const float*)d_in[13];
  const float* ps_w1 = (const float*)d_in[14];
  const float* ps_b1 = (const float*)d_in[15];
  const float* ps_w2 = (const float*)d_in[16];
  const float* ps_b2 = (const float*)d_in[17];
  const int*   assign= (const int*)d_in[18];
  const int*   hourp = (const int*)d_in[20];

  char* ws = (char*)d_ws;
  size_t off = 0;
  auto alloc = [&](size_t bytes) -> void* {
    void* p = ws + off;
    off = (off + bytes + 255) & ~(size_t)255;
    return p;
  };
  _Float16* emb_h  = (_Float16*)alloc((size_t)NB * EDIM * 2);
  _Float16* bigB   = (_Float16*)alloc((size_t)128 * 288 * 2);
  _Float16* B2h    = (_Float16*)alloc((size_t)128 * 64 * 2);
  float* hcat      = (float*)alloc((size_t)NB * 288 * 4);
  float* netArr    = (float*)alloc((size_t)NB * 4);
  int*   cidx      = (int*)alloc((size_t)NB * 4);
  int*   krank     = (int*)alloc((size_t)NB * 4);
  int*   gmap      = (int*)alloc((size_t)NB * 4);
  float* ws_flows  = (float*)alloc((size_t)NCL * KPC * KPC * 4);
  float* ws_eff    = (float*)alloc((size_t)NCL * KPC * KPC * 4);
  float* ws_tot    = (float*)alloc((size_t)NCL * 4);

  float* out     = (float*)d_out;
  float* outSent = out + 2 * NNsz + 1;
  float* outRecv = outSent + NB;
  float* outNet  = outRecv + NB;

  prep_kernel<<<NB / 256, 256, 0, stream>>>(assign, gen, cons, hourp, cidx, krank, gmap, netArr);
  cvt_kernel<<<(NB * EDIM + 128 * 288 + 128 * 64) / 256, 256, 0, stream>>>(
      emb, fp_w1, ps_w1, fp_w2, emb_h, bigB, B2h);
  gemm_hcat_kernel<<<(256 * 18) / 8, 256, 0, stream>>>(emb_h, bigB, hcat);
  cluster_kernel<<<NCL, 256, 0, stream>>>(gmap, netArr, hcat, pos, hourp,
      ps_b1, ps_w2, ps_b2, en_w1, en_b1, en_w2, en_b2,
      fp_w1, fp_b1, fp_b2, fp_w3, fp_b3,
      B2h, ws_flows, ws_eff, ws_tot, outSent, outRecv, outNet);
  fill_kernel<<<(unsigned)(NNsz / 4 / 256), 256, 0, stream>>>(cidx, krank, ws_flows, ws_eff, out);
  total_kernel<<<1, 64, 0, stream>>>(ws_tot, out);
}
